// SelfAttention_19542101197402
// MI455X (gfx1250) — compile-verified, matrix operand order checked by reading, not run on hardware
//
#include <hip/hip_runtime.h>
#include <hip/hip_bf16.h>
#include <math.h>

typedef __attribute__((ext_vector_type(16))) _Float16 v16h;
typedef __attribute__((ext_vector_type(8)))  float    v8f;
typedef __attribute__((__vector_size__(4 * sizeof(int)))) int v4i;

#define SEQ 1024
#define DMODEL 1024
#define HD 64

#define AS1 __attribute__((address_space(1)))
#define AS3 __attribute__((address_space(3)))

#if defined(__has_builtin)
#if __has_builtin(__builtin_amdgcn_global_load_async_to_lds_b128)
#define HAVE_ASYNC 1
#endif
#endif

#ifdef HAVE_ASYNC
#if __has_builtin(__builtin_amdgcn_s_wait_asynccnt)
#define ASYNC_WAIT(n) __builtin_amdgcn_s_wait_asynccnt(n)
#else
#define ASYNC_WAIT(n) asm volatile("s_wait_asynccnt " #n ::: "memory")
#endif
#else
#define ASYNC_WAIT(n)
#endif

// 16-byte tile-staging copy: async global->LDS (ASYNCcnt) when available,
// else synchronous through VGPRs.
__device__ __forceinline__ void copy16_to_lds(const _Float16* gsrc, _Float16* ldst) {
#ifdef HAVE_ASYNC
  __builtin_amdgcn_global_load_async_to_lds_b128(
      (AS1 v4i*)(gsrc), (AS3 v4i*)(ldst), 0, 0);
#else
  *(uint4*)ldst = *(const uint4*)gsrc;
#endif
}

__device__ __forceinline__ v8f wmma16(v16h a, v16h b, v8f c) {
  // v_wmma_f32_16x16x32_f16: D = A(16x32 f16) * B(32x16 f16) + C(16x16 f32)
  return __builtin_amdgcn_wmma_f32_16x16x32_f16(false, a, false, b, (short)0, c, false, false);
}

union FragU { v16h h; uint4 u[2]; };

// A/B fragment loader. Layout (ISA 7.12.2, 16-bit A 16x32):
//   lane L holds row (L&15); halves 0..7 = K in [koff, koff+8), halves 8..15 = K in [koff+16, koff+24)
//   where koff = 8*(L>>4). B mirrors A with "row" = output column of a [16][K] Bt tile.
__device__ __forceinline__ v16h load_frag(const _Float16* tile, int ldh) {
  const int lane = threadIdx.x & 31;
  const int row  = lane & 15;
  const int koff = (lane >> 4) << 3;
  const _Float16* p = tile + row * ldh + koff;
  FragU f;
  f.u[0] = *(const uint4*)(p);
  f.u[1] = *(const uint4*)(p + 16);
  return f.h;
}

__device__ __forceinline__ float wsum(float v) {
#pragma unroll
  for (int m = 16; m >= 1; m >>= 1) v += __shfl_xor(v, m, 32);
  return v;
}

// ---------------- conversion kernels ----------------
__global__ void cvt_f16(const float* __restrict__ in, _Float16* __restrict__ out, int n) {
  int i = blockIdx.x * blockDim.x + threadIdx.x;
  if (i < n) out[i] = (_Float16)in[i];
}

// W[K][N] fp32 -> Wt[N][K] f16  (Wt is the "Bt" operand: row = output column)
__global__ void cvt_f16_t(const float* __restrict__ W, _Float16* __restrict__ Wt, int K, int N) {
  int i = blockIdx.x * blockDim.x + threadIdx.x;
  if (i < K * N) {
    int nn = i / K, k = i - nn * K;
    Wt[i] = (_Float16)W[(size_t)k * N + nn];
  }
}

// ---------------- tiled WMMA GEMM: C[M][N] = A[M][K](f16) * Bt[N][K](f16)^T + bias ----------------
// Double-buffered LDS, async global->LDS staging overlapped with WMMA compute.
__global__ __launch_bounds__(256) void gemm_wmma(
    const _Float16* __restrict__ A, const _Float16* __restrict__ Bt,
    const float* __restrict__ bias, float* __restrict__ C, int M, int N, int K) {
  __shared__ _Float16 As[2][128 * 32];
  __shared__ _Float16 Bs[2][128 * 32];
  const int tid = threadIdx.x, wave = tid >> 5, lane = tid & 31;
  const int m0 = blockIdx.x * 128, n0 = blockIdx.y * 128;

  // Each thread stages 2 A-octets + 2 B-octets per K-step = 4 async-load instructions.
  auto issue_stage = [&](int buf, int k0) {
#pragma unroll
    for (int c = tid; c < 512; c += 256) {  // 128 rows x 4 octets of f16
      int r = c >> 2, s = c & 3;
      copy16_to_lds(&A[(size_t)(m0 + r) * K + k0 + s * 8], &As[buf][r * 32 + s * 8]);
      copy16_to_lds(&Bt[(size_t)(n0 + r) * K + k0 + s * 8], &Bs[buf][r * 32 + s * 8]);
    }
  };

  const v8f vzero = {0.f, 0.f, 0.f, 0.f, 0.f, 0.f, 0.f, 0.f};
  v8f acc[8];
#pragma unroll
  for (int j = 0; j < 8; ++j) acc[j] = vzero;

  const int nsteps = K / 32;
  issue_stage(0, 0);
  for (int i = 0; i < nsteps; ++i) {
    const int cur = i & 1;
    if (i + 1 < nsteps) {
      issue_stage(cur ^ 1, (i + 1) * 32);  // prefetch next stage (overlaps compute)
      ASYNC_WAIT(4);                       // oldest 4 (current stage) complete; next 4 in flight
    } else {
      ASYNC_WAIT(0);
    }
    __syncthreads();
    v16h a = load_frag(&As[cur][wave * 16 * 32], 32);
    v16h bfr[8];
#pragma unroll
    for (int j = 0; j < 8; ++j) bfr[j] = load_frag(&Bs[cur][j * 16 * 32], 32);
#pragma unroll
    for (int j = 0; j < 8; ++j) acc[j] = wmma16(a, bfr[j], acc[j]);
    __syncthreads();  // all waves done reading buf[cur] before it is refilled
  }

  // D layout: lane holds col (lane&15); VGPR e is row e (lanes 0-15) / e+8 (lanes 16-31)
  const int ncol = lane & 15, half = lane >> 4;
#pragma unroll
  for (int j = 0; j < 8; ++j) {
    int gc = n0 + j * 16 + ncol;
    float bv = bias[gc];
#pragma unroll
    for (int e = 0; e < 8; ++e) {
      int gr = m0 + wave * 16 + e + 8 * half;
      C[(size_t)gr * N + gc] = acc[j][e] + bv;
    }
  }
}

// ---------------- per-head LayerNorm + reshape to WMMA-friendly layouts ----------------
// qkv[4096][3072] fp32 -> qh,kh [B*H][N][64] f16 (k row-major == Bt for Q*K^T),
//                         vth   [B*H][64][N] f16 (v transposed == Bt for P*V)
__global__ __launch_bounds__(256) void ln_split(
    const float* __restrict__ qkv,
    const float* __restrict__ qg, const float* __restrict__ qb,
    const float* __restrict__ kg, const float* __restrict__ kb,
    _Float16* __restrict__ qh, _Float16* __restrict__ kh, _Float16* __restrict__ vth) {
  int task = blockIdx.x * 8 + (threadIdx.x >> 5);  // task = row*16 + h, 65536 total
  int lane = threadIdx.x & 31;
  int row = task >> 4, h = task & 15;
  int b = row >> 10, n = row & 1023;
  const float* base = qkv + (size_t)row * 3072;
  int d0 = lane, d1 = lane + 32;

  float q0 = base[h * 64 + d0], q1 = base[h * 64 + d1];
  float k0 = base[1024 + h * 64 + d0], k1 = base[1024 + h * 64 + d1];
  float v0 = base[2048 + h * 64 + d0], v1 = base[2048 + h * 64 + d1];

  float qmu = wsum(q0 + q1) * (1.f / 64.f);
  float dq0 = q0 - qmu, dq1 = q1 - qmu;
  float qr = rsqrtf(wsum(dq0 * dq0 + dq1 * dq1) * (1.f / 64.f) + 1e-5f);

  float kmu = wsum(k0 + k1) * (1.f / 64.f);
  float dk0 = k0 - kmu, dk1 = k1 - kmu;
  float kr = rsqrtf(wsum(dk0 * dk0 + dk1 * dk1) * (1.f / 64.f) + 1e-5f);

  size_t bh = (size_t)(b * 16 + h);
  size_t qko = (bh * SEQ + n) * 64;
  qh[qko + d0] = (_Float16)(dq0 * qr * qg[d0] + qb[d0]);
  qh[qko + d1] = (_Float16)(dq1 * qr * qg[d1] + qb[d1]);
  kh[qko + d0] = (_Float16)(dk0 * kr * kg[d0] + kb[d0]);
  kh[qko + d1] = (_Float16)(dk1 * kr * kg[d1] + kb[d1]);
  vth[bh * (64 * SEQ) + (size_t)d0 * SEQ + n] = (_Float16)v0;
  vth[bh * (64 * SEQ) + (size_t)d1 * SEQ + n] = (_Float16)v1;
}

// ---------------- flash attention: block = (b*h, 64-query tile), 4 waves x 16 queries ----------------
__global__ __launch_bounds__(128) void flash_attn(
    const _Float16* __restrict__ qh, const _Float16* __restrict__ kh,
    const _Float16* __restrict__ vth, _Float16* __restrict__ aout) {
  __shared__ _Float16 ksm[32 * 64];      // 32 keys x 64 hd
  __shared__ _Float16 vts[64 * 32];      // 64 hd x 32 keys (v^T)
  __shared__ _Float16 psm[4][16 * 32];   // per-wave P tile (A-fragment source)
  __shared__ float ssm[4][16 * 33];      // per-wave S scratch (padded)
  __shared__ float bfac[4][16];          // per-row broadcast factors

  const int tid = threadIdx.x, wave = tid >> 5, lane = tid & 31;
  const int bh = blockIdx.x;                       // 0..63
  const int q0 = blockIdx.y * 64 + wave * 16;      // query tile base
  const int r = lane & 15, half = lane >> 4;

  const _Float16* qp = qh + (size_t)bh * SEQ * HD;
  const _Float16* kp = kh + (size_t)bh * SEQ * HD;
  const _Float16* vp = vth + (size_t)bh * HD * SEQ;

  // Q A-fragments held in registers for the whole pass (hd split into two K=32 chunks)
  v16h aq0 = load_frag(qp + (size_t)q0 * HD, HD);
  v16h aq1 = load_frag(qp + (size_t)q0 * HD + 32, HD);

  const v8f vzero = {0.f, 0.f, 0.f, 0.f, 0.f, 0.f, 0.f, 0.f};
  v8f acc[4];
#pragma unroll
  for (int t = 0; t < 4; ++t) acc[t] = vzero;
  float m = -INFINITY, l = 0.f;

  for (int kb = 0; kb < 32; ++kb) {
    __syncthreads();
#pragma unroll
    for (int c = tid; c < 256; c += 128) {  // K tile: 32 rows x 8 octets
      int rr = c >> 3, s = c & 7;
      copy16_to_lds(&kp[(size_t)(kb * 32 + rr) * 64 + s * 8], &ksm[rr * 64 + s * 8]);
    }
#pragma unroll
    for (int c = tid; c < 256; c += 128) {  // V^T tile: 64 rows x 4 octets
      int dd = c >> 2, s = c & 3;
      copy16_to_lds(&vp[(size_t)dd * SEQ + kb * 32 + s * 8], &vts[dd * 32 + s * 8]);
    }
    ASYNC_WAIT(0);
    __syncthreads();

    // S = (Q K^T) * scale : two 16x16 key sub-tiles, K(hd)=64 in two WMMAs each
    v8f s0 = vzero, s1 = vzero;
    {
      v16h b00 = load_frag(&ksm[0], 64);
      v16h b01 = load_frag(&ksm[32], 64);
      s0 = wmma16(aq0, b00, s0);
      s0 = wmma16(aq1, b01, s0);
      v16h b10 = load_frag(&ksm[16 * 64], 64);
      v16h b11 = load_frag(&ksm[16 * 64 + 32], 64);
      s1 = wmma16(aq0, b10, s1);
      s1 = wmma16(aq1, b11, s1);
    }
#pragma unroll
    for (int e = 0; e < 8; ++e) {
      int mm = e + 8 * half;
      ssm[wave][mm * 33 + r]      = s0[e] * 0.125f;  // 1/sqrt(64)
      ssm[wave][mm * 33 + 16 + r] = s1[e] * 0.125f;
    }
    __syncthreads();

    // online softmax: lane pair (r, r+16) owns query row r; each scans 16 of 32 keys
    float pm = -INFINITY;
#pragma unroll
    for (int c = 0; c < 16; ++c) pm = fmaxf(pm, ssm[wave][r * 33 + half * 16 + c]);
    pm = fmaxf(pm, __shfl_xor(pm, 16, 32));
    float mnew = fmaxf(m, pm);
    float al = __expf(m - mnew);
    float ps = 0.f;
#pragma unroll
    for (int c = 0; c < 16; ++c) {
      float p = __expf(ssm[wave][r * 33 + half * 16 + c] - mnew);
      ps += p;
      psm[wave][r * 32 + half * 16 + c] = (_Float16)p;
    }
    ps += __shfl_xor(ps, 16, 32);
    l = al * l + ps;
    m = mnew;
    if (half == 0) bfac[wave][r] = al;
    __syncthreads();

    // rescale running O and accumulate P*V (4 hd sub-tiles of 16)
    float af[8];
#pragma unroll
    for (int e = 0; e < 8; ++e) af[e] = bfac[wave][e + 8 * half];
    v16h ap = load_frag(&psm[wave][0], 32);
#pragma unroll
    for (int t = 0; t < 4; ++t) {
#pragma unroll
      for (int e = 0; e < 8; ++e) acc[t][e] *= af[e];
      v16h bv = load_frag(&vts[16 * t * 32], 32);
      acc[t] = wmma16(ap, bv, acc[t]);
    }
  }

  // finalize: divide by row sums, emit f16 [B,N,D] with head-concat columns
  if (half == 0) bfac[wave][r] = 1.f / l;
  __syncthreads();
  float lf[8];
#pragma unroll
  for (int e = 0; e < 8; ++e) lf[e] = bfac[wave][e + 8 * half];
  const int b = bh >> 4, h = bh & 15;
#pragma unroll
  for (int t = 0; t < 4; ++t) {
    int gc = h * 64 + t * 16 + r;
#pragma unroll
    for (int e = 0; e < 8; ++e) {
      int gr = b * SEQ + q0 + e + 8 * half;
      aout[(size_t)gr * DMODEL + gc] = (_Float16)(acc[t][e] * lf[e]);
    }
  }
}

extern "C" void kernel_launch(void* const* d_in, const int* in_sizes, int n_in,
                              void* d_out, int out_size, void* d_ws, size_t ws_size,
                              hipStream_t stream) {
  const float* x    = (const float*)d_in[0];
  const float* Wqkv = (const float*)d_in[1];
  const float* bqkv = (const float*)d_in[2];
  const float* qg   = (const float*)d_in[3];
  const float* qb   = (const float*)d_in[4];
  const float* kg   = (const float*)d_in[5];
  const float* kb   = (const float*)d_in[6];
  const float* Wo   = (const float*)d_in[7];
  const float* bo   = (const float*)d_in[8];
  float* out = (float*)d_out;

  char* ws = (char*)d_ws;
  size_t off = 0;
  auto alloc = [&](size_t bytes) -> char* {
    char* p = ws + off;
    off += (bytes + 255) & ~(size_t)255;
    return p;
  };
  _Float16* xh   = (_Float16*)alloc(4096ull * 1024 * 2);  // x in f16
  _Float16* wqt  = (_Float16*)alloc(3072ull * 1024 * 2);  // Wqkv^T f16
  _Float16* wot  = (_Float16*)alloc(1024ull * 1024 * 2);  // Wo^T f16
  float*    qkv  = (float*)   alloc(4096ull * 3072 * 4);  // x@Wqkv + b, fp32
  _Float16* qh   = (_Float16*)alloc(64ull * 1024 * 64 * 2);
  _Float16* kh   = (_Float16*)alloc(64ull * 1024 * 64 * 2);
  _Float16* vth  = (_Float16*)alloc(64ull * 1024 * 64 * 2);
  _Float16* aout = (_Float16*)alloc(4096ull * 1024 * 2);

  cvt_f16<<<(4194304 + 255) / 256, 256, 0, stream>>>(x, xh, 4194304);
  cvt_f16_t<<<(3145728 + 255) / 256, 256, 0, stream>>>(Wqkv, wqt, 1024, 3072);
  cvt_f16_t<<<(1048576 + 255) / 256, 256, 0, stream>>>(Wo, wot, 1024, 1024);
  gemm_wmma<<<dim3(32, 24), 256, 0, stream>>>(xh, wqt, bqkv, qkv, 4096, 3072, 1024);
  ln_split<<<8192, 256, 0, stream>>>(qkv, qg, qb, kg, kb, qh, kh, vth);
  flash_attn<<<dim3(64, 16), 128, 0, stream>>>(qh, kh, vth, aout);
  gemm_wmma<<<dim3(32, 8), 256, 0, stream>>>(aout, wot, bo, out, 4096, 1024, 1024);
}
